// RecurrentPositionalEmbedddings_66554813219277
// MI455X (gfx1250) — matrix operand order repaired and checked by reading, not compiled, not run on hardware
//
#include <hip/hip_runtime.h>
#include <math.h>

// Problem constants (B=32, T=1024, C=1024)
#define BB   32
#define TT   1024
#define CC   1024
#define C2   2048

#define NWG      64     // persistent grid: 64 WGs * 256 threads = 512 waves
#define THREADS  256

// Dynamic LDS: 128KB staging (z or y1, 32x2048 f16) + 8KB f32 partial buffer
#define ZSH_HALF   (BB * C2)                 // 65536 halves = 128 KiB
#define SMEM_BYTES (ZSH_HALF * 2 + 8 * 1024)

#define AS1 __attribute__((address_space(1)))
#define AS3 __attribute__((address_space(3)))

typedef __attribute__((ext_vector_type(16))) _Float16 v16h;
typedef __attribute__((ext_vector_type(8)))  _Float16 v8h;
typedef __attribute__((ext_vector_type(8)))  float    v8f;
typedef __attribute__((ext_vector_type(4)))  int      v4i;

union V16H { v16h v; v8h h[2]; };

__device__ __forceinline__ float gelu_exact(float v) {
    return 0.5f * v * (1.0f + erff(v * 0.70710678118654752440f));
}

// 16-byte global -> LDS copy, async (ASYNCcnt path) when available.
__device__ __forceinline__ void copy16_g2lds(const _Float16* gsrc, _Float16* ldst) {
#if __has_builtin(__builtin_amdgcn_global_load_async_to_lds_b128)
    __builtin_amdgcn_global_load_async_to_lds_b128(
        (AS1 v4i*)(_Float16*)gsrc, (AS3 v4i*)ldst, 0, 0);
#else
    *(v8h*)ldst = *(const v8h*)gsrc;
#endif
}

__device__ __forceinline__ void wait_async_then_barrier() {
#if __has_builtin(__builtin_amdgcn_s_wait_asynccnt)
    __builtin_amdgcn_s_wait_asynccnt(0);
#endif
    __syncthreads();
}

// ---------------- device-wide sense-reversing barrier ----------------------
__device__ __forceinline__ void grid_sync(unsigned* cnt, unsigned* gen, unsigned nwg) {
    __syncthreads();
    if (threadIdx.x == 0) {
        __threadfence();  // publish this WG's global stores device-wide
        unsigned g = __hip_atomic_load(gen, __ATOMIC_RELAXED, __HIP_MEMORY_SCOPE_AGENT);
        unsigned prev = __hip_atomic_fetch_add(cnt, 1u, __ATOMIC_ACQ_REL, __HIP_MEMORY_SCOPE_AGENT);
        if (prev == nwg - 1u) {
            __hip_atomic_store(cnt, 0u, __ATOMIC_RELAXED, __HIP_MEMORY_SCOPE_AGENT);
            __hip_atomic_store(gen, g + 1u, __ATOMIC_RELEASE, __HIP_MEMORY_SCOPE_AGENT);
        } else {
            while (__hip_atomic_load(gen, __ATOMIC_ACQUIRE, __HIP_MEMORY_SCOPE_AGENT) == g) {
                __builtin_amdgcn_s_sleep(2);
            }
        }
        __threadfence();
    }
    __syncthreads();
}

// ---------------- init kernels --------------------------------------------
__global__ void cvt_f32_to_f16(const float* __restrict__ src,
                               _Float16* __restrict__ dst, size_t n) {
    size_t i = (size_t)blockIdx.x * blockDim.x + threadIdx.x;
    size_t stride = (size_t)gridDim.x * blockDim.x;
    for (; i < n; i += stride) dst[i] = (_Float16)src[i];
}

__global__ void zero_f16(_Float16* __restrict__ p, size_t n) {
    size_t i = (size_t)blockIdx.x * blockDim.x + threadIdx.x;
    size_t stride = (size_t)gridDim.x * blockDim.x;
    for (; i < n; i += stride) p[i] = (_Float16)0.0f;
}

__global__ void zero_u32(unsigned* __restrict__ p, int n) {
    int i = blockIdx.x * blockDim.x + threadIdx.x;
    if (i < n) p[i] = 0u;
}

// ---------------- persistent recurrent kernel -----------------------------
// Per step t (all 512 waves active in GEMM phases):
//   phase1: y1 = gelu([h, x_t] @ WaH^T + ba)  (32 x 2048)
//           WG b: columns [b*32, b*32+32); waves = 2 ntile x 2 mtile x 2 khalf
//   phase2: y2 = y1 @ WbH^T + bb              (32 x 1024)
//           WG b: columns [b*16, b*16+16); waves = 2 mtile x 4 kquarter
//   phase3: h' = LN(y2)*gamma+beta + y2 -> out[:,t,:], hH ; 32 waves
// WMMA operand mapping (wave32, v_wmma_f32_16x16x32_f16):
//   A (16x32 f16): lane L row m=L&15; halves 0..15 = K kb..kb+15,
//                  kb=(L>>4)*16   (from LDS)
//   B (32x16 f16): lane L col n=L&15; halves 0..15 = K bko..bko+15,
//                  bko=(L>>4)*16 (one 32B contiguous load from global/L2)
//   C/D (16x16 f32): vgpr r, lane L -> row r + 8*(L>>4), col L&15
__global__ void __launch_bounds__(THREADS)
rnn_persistent(const _Float16* __restrict__ WaH,   // 2048 x 2048 [n][k]
               const _Float16* __restrict__ WbH,   // 1024 x 2048 [n][k]
               const float* __restrict__ ba,       // 2048
               const float* __restrict__ bb,       // 1024
               const float* __restrict__ gamma,    // 1024
               const float* __restrict__ beta,     // 1024
               const _Float16* __restrict__ xH,    // (B,T,C) f16
               _Float16* __restrict__ hH,          // 32 x 1024 f16
               _Float16* __restrict__ y1H,         // 32 x 2048 f16
               float*    __restrict__ y2,          // 32 x 1024 f32
               float*    __restrict__ out,         // (B,T,C) f32
               unsigned* __restrict__ bar)         // [cnt, gen]
{
    extern __shared__ char smem[];
    _Float16* zsh  = (_Float16*)smem;                    // 32 x 2048 f16
    float*    psum = (float*)(smem + ZSH_HALF * 2);      // 8 waves x 256 f32

    const int lane = threadIdx.x & 31;
    const int wv   = threadIdx.x >> 5;                   // 0..7
    const int gw   = blockIdx.x * (THREADS / 32) + wv;   // 0..511
    const int ln15 = lane & 15;
    const int hi16 = lane >> 4;          // 0 or 1
    const int kb   = hi16 << 4;          // A sub-offset: 0 or 16 halves
    const int bko  = hi16 << 4;          // B sub-offset: 0 or 16 halves

    unsigned* cnt = bar;
    unsigned* gen = bar + 1;

    for (int t = 0; t < TT; ++t) {
        // ===== phase 1: stage z = [h | x_t] into LDS (async), then GEMM ====
        for (int c = threadIdx.x; c < ZSH_HALF / 8; c += THREADS) {  // 16B chunks
            const int row  = c >> 8;               // 256 chunks per row
            const int col8 = (c & 255) << 3;       // half offset in row
            const _Float16* src = (col8 < CC)
                ? (hH + (size_t)row * CC + col8)
                : (xH + ((size_t)row * TT + t) * CC + (col8 - CC));
            copy16_g2lds(src, zsh + (size_t)row * C2 + col8);
        }
        wait_async_then_barrier();

        {
            const int ntl   = wv & 1;                  // local N tile
            const int mtile = (wv >> 1) & 1;           // M tile (16 rows)
            const int khalf = wv >> 2;                 // K half (1024)
            const int bn    = (blockIdx.x * 2 + ntl) * 16 + ln15;
            const int arow  = mtile * 16 + ln15;
            const _Float16* zrow = zsh + (size_t)arow * C2;
            const _Float16* brow = WaH + (size_t)bn * C2;
            const int kbeg = khalf * 1024;

            v8f acc = {};
            #pragma unroll 2
            for (int k0 = kbeg; k0 < kbeg + 1024; k0 += 32) {
                V16H a, b;
                a.h[0] = *(const v8h*)(zrow + k0 + kb);        // ds_load_b128
                a.h[1] = *(const v8h*)(zrow + k0 + kb + 8);
                b.v    = *(const v16h*)(brow + k0 + bko);      // global x2 b128
                __builtin_prefetch((const void*)(brow + k0 + 64 + bko), 0, 1);
                acc = __builtin_amdgcn_wmma_f32_16x16x32_f16(
                          false, a.v, false, b.v, (short)0, acc, false, false);
            }
            #pragma unroll
            for (int r = 0; r < 8; ++r) psum[wv * 256 + r * 32 + lane] = acc[r];
            __syncthreads();
            if (khalf == 0) {   // waves 0..3 combine with waves 4..7
                const float bav = ba[bn];
                #pragma unroll
                for (int r = 0; r < 8; ++r) {
                    const float v = psum[wv * 256 + r * 32 + lane]
                                  + psum[(wv + 4) * 256 + r * 32 + lane] + bav;
                    const int m = mtile * 16 + r + (hi16 << 3);
                    y1H[(size_t)m * C2 + bn] = (_Float16)gelu_exact(v);
                }
            }
        }
        grid_sync(cnt, gen, NWG);

        // ===== phase 2: stage y1 into LDS (async), then GEMM ===============
        for (int c = threadIdx.x; c < ZSH_HALF / 8; c += THREADS)
            copy16_g2lds(y1H + (size_t)c * 8, zsh + (size_t)c * 8);
        wait_async_then_barrier();

        {
            const int mtile = wv & 1;                  // M tile
            const int kq    = wv >> 1;                 // K quarter (512)
            const int bn    = blockIdx.x * 16 + ln15;  // 64 WGs = 64 N tiles
            const int arow  = mtile * 16 + ln15;
            const _Float16* zrow = zsh + (size_t)arow * C2;
            const _Float16* brow = WbH + (size_t)bn * C2;
            const int kbeg = kq * 512;

            v8f acc = {};
            #pragma unroll 2
            for (int k0 = kbeg; k0 < kbeg + 512; k0 += 32) {
                V16H a, b;
                a.h[0] = *(const v8h*)(zrow + k0 + kb);
                a.h[1] = *(const v8h*)(zrow + k0 + kb + 8);
                b.v    = *(const v16h*)(brow + k0 + bko);
                __builtin_prefetch((const void*)(brow + k0 + 64 + bko), 0, 1);
                acc = __builtin_amdgcn_wmma_f32_16x16x32_f16(
                          false, a.v, false, b.v, (short)0, acc, false, false);
            }
            #pragma unroll
            for (int r = 0; r < 8; ++r) psum[wv * 256 + r * 32 + lane] = acc[r];
            __syncthreads();
            if (kq == 0) {      // waves 0,1 combine quarters {0,1,2,3}
                const float bbv = bb[bn];
                #pragma unroll
                for (int r = 0; r < 8; ++r) {
                    const int i = r * 32 + lane;
                    const float v = psum[(mtile + 0) * 256 + i]
                                  + psum[(mtile + 2) * 256 + i]
                                  + psum[(mtile + 4) * 256 + i]
                                  + psum[(mtile + 6) * 256 + i] + bbv;
                    const int m = mtile * 16 + r + (hi16 << 3);
                    y2[(size_t)m * CC + bn] = v;
                }
            }
        }
        grid_sync(cnt, gen, NWG);

        // ===== phase 3: LayerNorm + residual ===============================
        if (gw < BB) {
            const int b = gw;
            const float* row = y2 + (size_t)b * CC;
            float s = 0.0f, ss = 0.0f;
            for (int j = lane; j < CC; j += 32) {
                const float v = row[j];
                s += v; ss += v * v;
            }
            #pragma unroll
            for (int off = 16; off > 0; off >>= 1) {
                s  += __shfl_xor(s,  off, 32);
                ss += __shfl_xor(ss, off, 32);
            }
            const float mean = s * (1.0f / CC);
            const float var  = ss * (1.0f / CC) - mean * mean;
            const float rstd = rsqrtf(var + 1e-5f);
            float* orow = out + ((size_t)b * TT + t) * CC;
            _Float16* hrow = hH + (size_t)b * CC;
            for (int j = lane; j < CC; j += 32) {
                const float v  = row[j];
                const float hn = (v - mean) * rstd * gamma[j] + beta[j] + v;
                orow[j] = hn;
                hrow[j] = (_Float16)hn;
            }
        }
        grid_sync(cnt, gen, NWG);  // h' published before next step's staging
    }
}

// ---------------- host side ------------------------------------------------
extern "C" void kernel_launch(void* const* d_in, const int* in_sizes, int n_in,
                              void* d_out, int out_size, void* d_ws, size_t ws_size,
                              hipStream_t stream) {
    const float* x     = (const float*)d_in[0];  // (32,1024,1024)
    const float* Wa    = (const float*)d_in[1];  // (2048,2048)
    const float* ba    = (const float*)d_in[2];  // (2048)
    const float* Wb    = (const float*)d_in[3];  // (1024,2048)
    const float* bb    = (const float*)d_in[4];  // (1024)
    const float* gamma = (const float*)d_in[5];  // (1024)
    const float* beta  = (const float*)d_in[6];  // (1024)
    float* out = (float*)d_out;

    char* ws = (char*)d_ws;
    const size_t WA_H  = 0;                                  //  8 MiB
    const size_t WB_H  = WA_H + (size_t)C2 * C2 * 2;         //  4 MiB
    const size_t X_H   = WB_H + (size_t)CC * C2 * 2;         // 64 MiB
    const size_t H_H   = X_H  + (size_t)BB * TT * CC * 2;    // 64 KiB
    const size_t Y1_H  = H_H  + (size_t)BB * CC * 2;         // 128 KiB
    const size_t Y2_F  = Y1_H + (size_t)BB * C2 * 2;         // 128 KiB
    const size_t BAR   = Y2_F + (size_t)BB * CC * 4;

    _Float16* WaH = (_Float16*)(ws + WA_H);
    _Float16* WbH = (_Float16*)(ws + WB_H);
    _Float16* xH  = (_Float16*)(ws + X_H);
    _Float16* hH  = (_Float16*)(ws + H_H);
    _Float16* y1H = (_Float16*)(ws + Y1_H);
    float*    y2  = (float*)   (ws + Y2_F);
    unsigned* bar = (unsigned*)(ws + BAR);

    cvt_f32_to_f16<<<1024, 256, 0, stream>>>(Wa, WaH, (size_t)C2 * C2);
    cvt_f32_to_f16<<<1024, 256, 0, stream>>>(Wb, WbH, (size_t)CC * C2);
    cvt_f32_to_f16<<<4096, 256, 0, stream>>>(x,  xH,  (size_t)BB * TT * CC);
    zero_f16<<<64, 256, 0, stream>>>(hH, (size_t)BB * CC);       // h0 = 0
    zero_u32<<<1, 32, 0, stream>>>(bar, 8);                      // barrier state

    rnn_persistent<<<NWG, THREADS, SMEM_BYTES, stream>>>(
        WaH, WbH, ba, bb, gamma, beta, xH, hH, y1H, y2, out, bar);
}
